// QuaternionLinear_87462714016219
// MI455X (gfx1250) — matrix operand order repaired and checked by reading, not compiled, not run on hardware
//
#include <hip/hip_runtime.h>

// CDNA5 (gfx1250) wave32 WMMA types
typedef __attribute__((ext_vector_type(16))) _Float16 v16h;
typedef __attribute__((ext_vector_type(8)))  float    v8f;
typedef __attribute__((ext_vector_type(4)))  float    v4f;
typedef __attribute__((ext_vector_type(4)))  unsigned int v4u;
typedef __attribute__((ext_vector_type(4)))  int      v4i;

#define K_DIM 256
#define N_DIM 256
#define SLICE_HALVES 16384            // one K-step slice: 16 hi frags + 16 lo frags = 32 KB
#define LO_OFF       8192             // lo fragments start at +16 KB within a slice
#define BLOCK_THREADS 512             // 16 wave32 -> 256 M rows per block

// ---------------------------------------------------------------------------
// Async global->LDS copy (16B per lane), ASYNCcnt-tracked; falls back to
// load+ds_store if the builtin is unavailable.
// ---------------------------------------------------------------------------
#if defined(__has_builtin)
#if __has_builtin(__builtin_amdgcn_global_load_async_to_lds_b128)
#define HAVE_ASYNC_LDS 1
#endif
#endif

__device__ __forceinline__ void async_cp16(const _Float16* g, _Float16* l)
{
#ifdef HAVE_ASYNC_LDS
    __builtin_amdgcn_global_load_async_to_lds_b128(
        (__attribute__((address_space(1))) v4i*)(uintptr_t)g,
        (__attribute__((address_space(3))) v4i*)(unsigned int)(uintptr_t)l,
        0, 0);
#else
    *(v4u*)l = *(const v4u*)g;        // global_load_b128 + ds_store_b128
#endif
}

__device__ __forceinline__ void wait_async()
{
#ifdef HAVE_ASYNC_LDS
#if __has_builtin(__builtin_amdgcn_s_wait_asynccnt)
    __builtin_amdgcn_s_wait_asynccnt(0);
#else
    asm volatile("s_wait_asynccnt 0" ::: "memory");
#endif
#endif
}

// stage one 32 KB K-slice with all 512 threads (4 x 16B per thread)
__device__ __forceinline__ void stage_slice(const _Float16* __restrict__ src,
                                            _Float16* dst, int tid)
{
#pragma unroll
    for (int i = 0; i < 4; ++i) {
        const int off = (i * BLOCK_THREADS + tid) * 8;   // in halves, 16B chunks
        async_cp16(src + off, dst + off);
    }
}

// ---------------------------------------------------------------------------
// Kernel A: build effective quaternion weight matrix Weff[256x256], f16 hi/lo
// split, stored K-slice-major in CDNA5 WMMA B-fragment layout:
//   ws[kstep*16384 + ntile*512 + lane*16 + pos]            (hi)
//   ws[kstep*16384 + 8192 + ntile*512 + lane*16 + pos]     (lo)
// B fragment (32Kx16N f16, wave32, contiguous k order used by this code):
//   lanes 0-15 : N=lane,    K = kstep*32 + (0..15)  -> halves 0..15
//   lanes16-31 : N=lane-16, K = kstep*32 + (16..31) -> halves 0..15
// ---------------------------------------------------------------------------
__global__ __launch_bounds__(256)
void build_weff_frag(const float* __restrict__ Wr, const float* __restrict__ Wi,
                     const float* __restrict__ Wj, const float* __restrict__ Wk,
                     _Float16* __restrict__ ws)
{
    const int idx = blockIdx.x * 256 + threadIdx.x;   // 0..65535
    const int k = idx >> 8;      // input dim  (K)
    const int j = idx & 255;     // output dim (N)
    const int n = k >> 2, c = k & 3;
    const int m = j >> 2, d = j & 3;

    // Hamilton product table: W4[d][c] = (component, sign)
    const int   comp_tbl[4][4] = {{0,1,2,3},{1,0,3,2},{2,3,0,1},{3,2,1,0}};
    const float sign_tbl[4][4] = {{ 1.f,-1.f,-1.f,-1.f},
                                  { 1.f, 1.f, 1.f,-1.f},
                                  { 1.f,-1.f, 1.f, 1.f},
                                  { 1.f, 1.f,-1.f, 1.f}};
    const float* Wc[4] = {Wr, Wi, Wj, Wk};

    const float w  = sign_tbl[d][c] * Wc[comp_tbl[d][c]][m * 64 + n];
    const _Float16 hi = (_Float16)w;
    const _Float16 lo = (_Float16)(w - (float)hi);

    const int kstep = k >> 5, kk = k & 31;
    const int ntile = j >> 4, n16 = j & 15;
    const int lane  = (kk < 16) ? n16 : (16 + n16);
    const int pos   = kk & 15;
    const int hbase = kstep * SLICE_HALVES + ntile * 512 + lane * 16 + pos;

    ws[hbase]          = hi;
    ws[hbase + LO_OFF] = lo;
}

// ---------------------------------------------------------------------------
// Kernel B: GEMM  out[M,256] = X[M,256] @ Weff[256,256]
// Block = 16 wave32 = 256 M rows; each wave: 16-row M tile x all 16 N tiles.
// B double-buffered in LDS (2 x 32KB slices), staged with ASYNCcnt-tracked
// global_load_async_to_lds while the previous slice is consumed.
// A stream is software-pipelined one K-step ahead to hide HBM latency across
// the workgroup barrier. fp32-grade accuracy via f16 hi/lo split.
// ---------------------------------------------------------------------------
__global__ __launch_bounds__(BLOCK_THREADS)
void qlinear_wmma(const float* __restrict__ x,
                  const _Float16* __restrict__ wfrag,
                  float* __restrict__ out)
{
    __shared__ _Float16 smem[2][SLICE_HALVES];   // 64 KB total

    const int tid  = threadIdx.x;
    const int lane = tid & 31;
    const int wave = tid >> 5;                       // 0..15
    const long row0 = (long)blockIdx.x * 256 + wave * 16;
    const int  r    = lane & 15;                     // A row in tile
    const int  koff = (lane >> 4) * 16;              // A K sub-offset
    const float* xrow = x + (row0 + r) * (long)K_DIM + koff;

    v8f acc[16];
#pragma unroll
    for (int t = 0; t < 16; ++t)
#pragma unroll
        for (int p = 0; p < 8; ++p) acc[t][p] = 0.0f;

    // prologue: stage slice 0 and prefetch A(ks=0)
    stage_slice(wfrag, smem[0], tid);
    v4f a0 = *(const v4f*)(xrow + 0);
    v4f a1 = *(const v4f*)(xrow + 4);
    v4f a2 = *(const v4f*)(xrow + 8);
    v4f a3 = *(const v4f*)(xrow + 12);
    wait_async();
    __syncthreads();

#pragma unroll
    for (int ks = 0; ks < 8; ++ks) {
        // stage next B slice into the other buffer (overlaps with compute)
        if (ks < 7)
            stage_slice(wfrag + (ks + 1) * SLICE_HALVES, smem[(ks + 1) & 1], tid);

        // prefetch A(ks+1): a full iteration of WMMA work hides HBM latency
        v4f n0 = a0, n1 = a1, n2 = a2, n3 = a3;
        if (ks < 7) {
            const float* pn = xrow + (ks + 1) * 32;
            n0 = *(const v4f*)(pn + 0);
            n1 = *(const v4f*)(pn + 4);
            n2 = *(const v4f*)(pn + 8);
            n3 = *(const v4f*)(pn + 12);
        }

        // --- fp32 -> f16 hi/lo split of A(ks) (VALU co-executes with XDL) ---
        float av[16];
        *(v4f*)(av + 0)  = a0;
        *(v4f*)(av + 4)  = a1;
        *(v4f*)(av + 8)  = a2;
        *(v4f*)(av + 12) = a3;
        v16h ahi, alo;
#pragma unroll
        for (int i = 0; i < 16; ++i) {
            const float f = av[i];
            const _Float16 h = (_Float16)f;
            ahi[i] = h;
            alo[i] = (_Float16)(f - (float)h);
        }

        // --- 16 N-tiles, B fragments from LDS ---
        const _Float16* bs = smem[ks & 1];
#pragma unroll
        for (int t = 0; t < 16; ++t) {
            const int foff = t * 512 + lane * 16;
            const v16h bh = *(const v16h*)(bs + foff);
            const v16h bl = *(const v16h*)(bs + LO_OFF + foff);
            acc[t] = __builtin_amdgcn_wmma_f32_16x16x32_f16(
                false, ahi, false, bh, (short)0, acc[t], false, false);
            acc[t] = __builtin_amdgcn_wmma_f32_16x16x32_f16(
                false, alo, false, bh, (short)0, acc[t], false, false);
            acc[t] = __builtin_amdgcn_wmma_f32_16x16x32_f16(
                false, ahi, false, bl, (short)0, acc[t], false, false);
        }

        // staging of next slice complete; buffer reuse safe (this iteration's
        // ds reads were all consumed by the WMMAs above)
        wait_async();
        __syncthreads();

        a0 = n0; a1 = n1; a2 = n2; a3 = n3;
    }

    // --- store: C/D layout: VGPR p -> M = p + (lane>=16)*8, N = lane&15 ---
    const int outcol = lane & 15;
    const int rowadd = (lane >> 4) * 8;
#pragma unroll
    for (int t = 0; t < 16; ++t) {
#pragma unroll
        for (int p = 0; p < 8; ++p) {
            const long orow = row0 + rowadd + p;
            out[orow * (long)N_DIM + t * 16 + outcol] = acc[t][p];
        }
    }
}

// ---------------------------------------------------------------------------
extern "C" void kernel_launch(void* const* d_in, const int* in_sizes, int n_in,
                              void* d_out, int out_size, void* d_ws, size_t ws_size,
                              hipStream_t stream)
{
    const float* x  = (const float*)d_in[0];   // q_in    (16, 8192, 256) f32
    const float* wr = (const float*)d_in[1];   // r_weight (64, 64) f32
    const float* wi = (const float*)d_in[2];
    const float* wj = (const float*)d_in[3];
    const float* wk = (const float*)d_in[4];

    _Float16* ws = (_Float16*)d_ws;            // 256 KB, K-slice-major hi/lo

    // 1) build fragment-native effective weights (deterministic, every call)
    build_weff_frag<<<256, 256, 0, stream>>>(wr, wi, wj, wk, ws);

    // 2) WMMA GEMM: M=131072 rows, 256 rows per block (16 wave32)
    const int M = 16 * 8192;
    qlinear_wmma<<<M / 256, BLOCK_THREADS, 0, stream>>>(x, ws, (float*)d_out);
}
